// node2vec_rnn_16140487098664
// MI455X (gfx1250) — matrix operand order repaired and checked by reading, not compiled, not run on hardware
//
#include <hip/hip_runtime.h>
#include <math.h>

// ---------------------------------------------------------------------------
// node2vec GRU + sampled loss for MI455X (gfx1250), wave32, WMMA f16->f32.
//
// V=500000, E=128, H=128, L=10, P=10, N=50  ->  batch B = 1+10+50 = 61 (pad 64)
//
// Kernel 1: gather embedding rows + convert x / w_ih / w_hh to f16 in d_ws.
// Kernel 2: single workgroup (16 waves). Per GRU step:
//   - x_t tile (64x128 f16) staged into LDS with GLOBAL_LOAD_ASYNC_TO_LDS_B128
//     (ASYNCcnt + s_wait_asynccnt), one L2 read per step.
//   - 64x384 gate pre-activations via v_wmma_f32_16x16x32_f16
//     (96 tile jobs x 8 WMMA), A-operands from LDS, weights from L2.
//   - gates + h update with h carried in registers; final loss on-chip.
// ---------------------------------------------------------------------------

#define EE      128
#define HHH     128
#define LL      10
#define PP      10
#define NNEG    50
#define BREAL   61
#define BPAD    64
#define GG      384          // 3*H
#define THREADS 512
#define WAVES   16

typedef __attribute__((ext_vector_type(16))) _Float16 v16h;
typedef __attribute__((ext_vector_type(8)))  _Float16 v8h;
typedef __attribute__((ext_vector_type(8)))  float    v8f;

// ---------------------------------------------------------------------------
// Kernel 1: gather + f32->f16 conversion into workspace.
// ---------------------------------------------------------------------------
__global__ void gather_cvt_kernel(const int*   __restrict__ phr,
                                  const int*   __restrict__ pos,
                                  const int*   __restrict__ neg,
                                  const float* __restrict__ u_emb,
                                  const float* __restrict__ v_emb,
                                  const float* __restrict__ w_ih,
                                  const float* __restrict__ w_hh,
                                  _Float16*    __restrict__ xh,
                                  _Float16*    __restrict__ wih_h,
                                  _Float16*    __restrict__ whh_h)
{
    const int i  = blockIdx.x * blockDim.x + threadIdx.x;
    const int NX = BPAD * LL * EE;   // 81920 x elements (incl. zero padding rows)
    const int NW = GG * EE;          // 49152 weight elements per matrix

    if (i < NX) {
        const int e = i & (EE - 1);
        const int l = (i >> 7) % LL;
        const int b = i / (LL * EE);
        float v = 0.0f;
        if (b == 0) {
            v = u_emb[(size_t)phr[l] * EE + e];
        } else if (b <= PP) {
            v = v_emb[(size_t)pos[(b - 1) * LL + l] * EE + e];
        } else if (b < BREAL) {
            v = v_emb[(size_t)neg[(b - 1 - PP) * LL + l] * EE + e];
        }
        xh[i] = (_Float16)v;         // layout: ((b*L + l)*E + e)
    } else {
        int j = i - NX;
        if (j < NW) {
            wih_h[j] = (_Float16)w_ih[j];
        } else {
            j -= NW;
            if (j < NW) whh_h[j] = (_Float16)w_hh[j];
        }
    }
}

// ---------------------------------------------------------------------------
// WMMA operand loaders (CDNA5 16-bit layouts, ISA 7.12.2).
// A (16x32, MxK): lane<16 -> M=lane,  K = {k0..k0+7} U {k0+16..k0+23}
//                 lane>=16 -> M=lane-16, K shifted by +8 / +24.
// B (32x16, KxN): lane<16 -> N=lane,  K = {k0..k0+7} U {k0+16..k0+23}
//                 lane>=16 -> N=lane-16, K shifted by +8 / +24.
// Both resolve to contiguous 16B chunks of the row-major source rows.
// ---------------------------------------------------------------------------
__device__ __forceinline__ v16h load_a_tile(const _Float16* row, int k0, int lane) {
    const int kb = k0 + ((lane & 16) >> 1);   // +8 halves for the upper half-wave
    v8h lo = *reinterpret_cast<const v8h*>(row + kb);
    v8h hi = *reinterpret_cast<const v8h*>(row + kb + 16);
    return __builtin_shufflevector(lo, hi, 0,1,2,3,4,5,6,7,8,9,10,11,12,13,14,15);
}

__device__ __forceinline__ v16h load_b_tile(const _Float16* row, int k0, int lane) {
    const int kb = k0 + ((lane & 16) >> 1);   // +8 halves for the upper half-wave
    v8h lo = *reinterpret_cast<const v8h*>(row + kb);
    v8h hi = *reinterpret_cast<const v8h*>(row + kb + 16);
    return __builtin_shufflevector(lo, hi, 0,1,2,3,4,5,6,7,8,9,10,11,12,13,14,15);
}

__device__ __forceinline__ v8f wmma_f16(v16h a, v16h b, v8f c) {
    // (neg_a, A, neg_b, B, c_mod, C, reuse_a, reuse_b)
    return __builtin_amdgcn_wmma_f32_16x16x32_f16(false, a, false, b,
                                                  (short)0, c, false, false);
}

// D (16x16 f32): element r of v8f, lane c<16 -> (M=r,   N=c)
//                              lane c>=16    -> (M=8+r, N=c-16)
__device__ __forceinline__ void store_c_tile(float* dst, int ldc,
                                             int m0, int c0, int lane, v8f acc) {
    const int rbase = m0 + ((lane & 16) >> 1);
    const int col   = c0 + (lane & 15);
#pragma unroll
    for (int r = 0; r < 8; ++r)
        dst[(rbase + r) * ldc + col] = acc[r];
}

// ---------------------------------------------------------------------------
// Kernel 2: single-workgroup GRU (10 steps) + loss.  Static LDS = 160 KB.
// ---------------------------------------------------------------------------
__global__ __launch_bounds__(THREADS)
void gru_loss_kernel(const _Float16* __restrict__ xh,
                     const _Float16* __restrict__ wih,   // (384,128) f16 row-major
                     const _Float16* __restrict__ whh,   // (384,128) f16 row-major
                     const float*    __restrict__ b_ih,
                     const float*    __restrict__ b_hh,
                     const float*    __restrict__ h0,
                     float*          __restrict__ out)
{
    __shared__ __align__(16) _Float16 sH16[BPAD * HHH];     // 16 KB (WMMA A: h)
    __shared__ __align__(16) _Float16 sX16[BPAD * HHH];     // 16 KB (WMMA A: x_t)
    __shared__ float sGrz[BPAD * 256];                      // 64 KB (r,z preact sum)
    __shared__ float sGni[BPAD * HHH];                      // 32 KB (x-side n preact)
    __shared__ float sGnh[BPAD * HHH];                      // 32 KB (h-side n preact)

    const int tid  = threadIdx.x;
    const int lane = tid & 31;
    const int wv   = tid >> 5;

    // Hidden state: 16 (b,h) pairs per thread, f32 in registers.
    float hreg[16];
#pragma unroll
    for (int j = 0; j < 16; ++j) {
        const int p = tid + THREADS * j;        // p = b*128 + h, covers 0..8191
        const float v = h0[p & (HHH - 1)];
        hreg[j] = v;
        sH16[p] = (_Float16)v;
    }
    __syncthreads();

    for (int t = 0; t < LL; ++t) {
        // --- Async-stage x_t (64x128 f16, rows strided by L*E) into LDS -------
        // 1024 16-byte chunks; 2 per thread, ASYNCcnt-tracked.
#pragma unroll
        for (int k = 0; k < 2; ++k) {
            const int c   = tid + THREADS * k;           // chunk id 0..1023
            const int b   = c >> 3;                      // 8 chunks per row
            const int sub = c & 7;
            const uint64_t gaddr =
                (uint64_t)(uintptr_t)(xh + ((size_t)b * LL + t) * EE + sub * 8);
            const uint32_t laddr = (uint32_t)(uintptr_t)(&sX16[c * 8]);
            asm volatile("global_load_async_to_lds_b128 %0, %1, off"
                         :: "v"(laddr), "v"(gaddr)
                         : "memory");
        }
        asm volatile("s_wait_asynccnt 0x0" ::: "memory");
        __syncthreads();

        // --- GEMM phase: 96 tile jobs of 16x16 output, K=128 ------------------
        for (int job = wv; job < 96; job += WAVES) {
            const int mt = job & 3;
            const int nt = job >> 2;
            const int m0 = mt << 4;
            const int g0 = nt << 4;

            const int arow = m0 + (lane & 15);
            const _Float16* xrow   = &sX16[arow * HHH];
            const _Float16* hrow   = &sH16[arow * HHH];
            const int bn = g0 + (lane & 15);
            const _Float16* wihrow = wih + (size_t)bn * EE;
            const _Float16* whhrow = whh + (size_t)bn * EE;

            if (nt < 16) {
                // r/z gates: x@Wih^T + h@Whh^T fused into one accumulator.
                v8f c = {};
#pragma unroll
                for (int k0 = 0; k0 < EE; k0 += 32) {
                    c = wmma_f16(load_a_tile(xrow, k0, lane),
                                 load_b_tile(wihrow, k0, lane), c);
                    c = wmma_f16(load_a_tile(hrow, k0, lane),
                                 load_b_tile(whhrow, k0, lane), c);
                }
                store_c_tile(sGrz, 256, m0, g0, lane, c);
            } else {
                // n gate: keep the two matmuls separate (n = tanh(in + r*hn)).
                v8f ci = {}, ch = {};
#pragma unroll
                for (int k0 = 0; k0 < EE; k0 += 32) {
                    ci = wmma_f16(load_a_tile(xrow, k0, lane),
                                  load_b_tile(wihrow, k0, lane), ci);
                    ch = wmma_f16(load_a_tile(hrow, k0, lane),
                                  load_b_tile(whhrow, k0, lane), ch);
                }
                store_c_tile(sGni, HHH, m0, g0 - 256, lane, ci);
                store_c_tile(sGnh, HHH, m0, g0 - 256, lane, ch);
            }
        }
        __syncthreads();

        // --- Gate phase: elementwise update of h ------------------------------
#pragma unroll
        for (int j = 0; j < 16; ++j) {
            const int p  = tid + THREADS * j;
            const int b  = p >> 7;
            const int hh = p & (HHH - 1);
            const float rr = sGrz[b * 256 + hh]       + b_ih[hh]         + b_hh[hh];
            const float zz = sGrz[b * 256 + 128 + hh] + b_ih[128 + hh]   + b_hh[128 + hh];
            const float r  = 1.0f / (1.0f + expf(-rr));
            const float z  = 1.0f / (1.0f + expf(-zz));
            const float nn = tanhf(sGni[p] + b_ih[256 + hh] +
                                   r * (sGnh[p] + b_hh[256 + hh]));
            const float hn = (1.0f - z) * nn + z * hreg[j];
            hreg[j] = hn;
            sH16[p] = (_Float16)hn;
        }
        __syncthreads();
    }

    // --- Loss tail: stage final h (f32) in sGni; sGrz[0..63] holds the dots ---
#pragma unroll
    for (int j = 0; j < 16; ++j)
        sGni[tid + THREADS * j] = hreg[j];
    __syncthreads();

    if (tid >= 1 && tid < BREAL) {
        float d = 0.0f;
        for (int h = 0; h < HHH; ++h)
            d += sGni[h] * sGni[tid * HHH + h];
        sGrz[tid] = d * (1.0f / (float)HHH);
    }
    __syncthreads();

    if (tid == 0) {
        float pos_loss = 0.0f;
        for (int b = 1; b <= PP; ++b)
            pos_loss += sGrz[b];
        float neg_sum = 0.0f;
        for (int b = 1 + PP; b < BREAL; ++b) {
            const float s = sGrz[b];
            neg_sum += (s > 0.0f) ? expf(s) : 0.0f;
        }
        // -(neg_loss + pos_loss) with neg_loss = -log(1 + neg_sum)
        out[0] = logf(1.0f + neg_sum) - pos_loss;
    }
}

// ---------------------------------------------------------------------------
// Launch wrapper.
// ---------------------------------------------------------------------------
extern "C" void kernel_launch(void* const* d_in, const int* in_sizes, int n_in,
                              void* d_out, int out_size, void* d_ws, size_t ws_size,
                              hipStream_t stream) {
    const int*   phr   = (const int*)  d_in[0];
    const int*   pos   = (const int*)  d_in[1];
    const int*   neg   = (const int*)  d_in[2];
    const float* u_emb = (const float*)d_in[3];
    const float* v_emb = (const float*)d_in[4];
    const float* w_ih  = (const float*)d_in[5];
    const float* w_hh  = (const float*)d_in[6];
    const float* b_ih  = (const float*)d_in[7];
    const float* b_hh  = (const float*)d_in[8];
    const float* h0    = (const float*)d_in[9];
    float* out = (float*)d_out;

    // Workspace carve-up (f16): x (64*10*128) | w_ih (384*128) | w_hh (384*128)
    _Float16* xh    = (_Float16*)d_ws;
    _Float16* wih_h = xh + (size_t)BPAD * LL * EE;
    _Float16* whh_h = wih_h + (size_t)GG * EE;

    const int total  = BPAD * LL * EE + 2 * GG * EE;   // 180224
    const int blocks = (total + 255) / 256;            // 704

    gather_cvt_kernel<<<blocks, 256, 0, stream>>>(phr, pos, neg, u_emb, v_emb,
                                                  w_ih, w_hh, xh, wih_h, whh_h);
    gru_loss_kernel<<<1, THREADS, 0, stream>>>(xh, wih_h, whh_h,
                                               b_ih, b_hh, h0, out);
}
